// LSSMWithS4Parameterization_60069412601933
// MI455X (gfx1250) — compile-verified, matrix operand order checked by reading, not run on hardware
//
#include <hip/hip_runtime.h>

#define Hd 128
#define Bd 128
#define Ld 2048
#define NC 16
#define LC (Ld / NC)      // 128 steps per chunk
#define LT 32             // time tile inside main kernel
#define NTILES (LC / LT)  // 4
#define LDP 136           // padded LDS row length in halfs (breaks bank aliasing)

static constexpr float DT = 0.01f;

typedef __attribute__((ext_vector_type(16))) _Float16 v16h;
typedef __attribute__((ext_vector_type(8)))  float    v8f;

// ---------------------------------------------------------------------------
// Kernel 1: parameter setup.
//   Abar[i][j] = (1/(I - A*dt/2)) * (I + A*dt/2)        (elementwise, I=eye)
//   Cbar[i][j] = (1/(I - Abar*dt/2)) * C * dt           (elementwise)
//   Apow       = Abar^128 (per-element, for chunk prefix)
//   Cbar is written directly in WMMA B-fragment layout (f16), minus sign of
//   the imaginary part folded in:  Y = HRe@CbRe^T + HIm@(-CbIm^T)
// ---------------------------------------------------------------------------
__global__ void lssm_setup_kernel(const float* __restrict__ A_rt,
                                  const float* __restrict__ C_rt,
                                  float* __restrict__ abarRe,
                                  float* __restrict__ abarIm,
                                  float* __restrict__ apow,
                                  _Float16* __restrict__ bfrag) {
    const int i = blockIdx.x;    // row of Abar/Cbar
    const int j = threadIdx.x;   // col of Abar/Cbar
    const int idx = i * Hd + j;

    const float ar = A_rt[idx * 2 + 0], ai = A_rt[idx * 2 + 1];
    const float cr = C_rt[idx * 2 + 0], ci = C_rt[idx * 2 + 1];
    const float hdt = DT * 0.5f;
    const float eye = (i == j) ? 1.0f : 0.0f;

    // Abar = recip(I - z) * (I + z), z = A*dt/2
    const float zr = ar * hdt, zi = ai * hdt;
    float dr = eye - zr, di = -zi;
    float den = dr * dr + di * di;
    float rr = dr / den, ri = -di / den;
    const float nr = eye + zr, ni = zi;
    const float abr = rr * nr - ri * ni;
    const float abi = rr * ni + ri * nr;
    abarRe[idx] = abr;
    abarIm[idx] = abi;

    // Apow = Abar^128 via 7 complex squarings
    float pr = abr, pi = abi;
#pragma unroll
    for (int s = 0; s < 7; ++s) {
        float t = pr * pr - pi * pi;
        pi = 2.0f * pr * pi;
        pr = t;
    }
    apow[idx * 2 + 0] = pr;
    apow[idx * 2 + 1] = pi;

    // Cbar = recip(I - Abar*dt/2) * C * dt
    float er = eye - abr * hdt, ei = -abi * hdt;
    float den2 = er * er + ei * ei;
    float qr = er / den2, qi = -ei / den2;
    const float cbr = (qr * cr - qi * ci) * DT;
    const float cbi = (qr * ci + qi * cr) * DT;

    // B matrix element (k = j, n = i): B1[k][n] = CbRe[n][k], B2[k][n] = -CbIm[n][k]
    // Fragment layout (mirrors ISA 16-bit operand layout):
    //   kb = k>>5, kk = k&31, nb = n>>4, nn = n&15
    //   lane    = nn + 16*((kk>>3)&1)
    //   halfIdx = ((kk>>4)<<3) + (kk&7)
    const int kb = j >> 5, kk = j & 31;
    const int nb = i >> 4, nn = i & 15;
    const int lane = nn + (((kk >> 3) & 1) << 4);
    const int hidx = ((kk >> 4) << 3) + (kk & 7);
    const int base = lane * 16 + hidx;
    bfrag[(size_t)((kb * 8 + nb)) * 512 + base]      = (_Float16)cbr;   // Re plane
    bfrag[(size_t)(32 + kb * 8 + nb) * 512 + base]   = (_Float16)(-cbi); // -Im plane
}

// ---------------------------------------------------------------------------
// Kernel 2: per-chunk carry.  Scan each 128-step chunk from zero state.
// carry[b][c][h] = h-state at end of chunk c given zero start state.
// ---------------------------------------------------------------------------
__global__ void lssm_carry_kernel(const float* __restrict__ X,
                                  const float* __restrict__ abarRe,
                                  const float* __restrict__ abarIm,
                                  float* __restrict__ carry) {
    const int c = blockIdx.x;
    const int b = blockIdx.y;
    const int h = threadIdx.x;
    const float ar = abarRe[b * Hd + h];
    const float ai = abarIm[b * Hd + h];
    const float* xp = X + ((size_t)b * Ld + (size_t)c * LC) * Hd + h;
    float hr = 0.0f, hi = 0.0f;
#pragma unroll 4
    for (int t = 0; t < LC; ++t) {
        const float x = xp[(size_t)t * Hd];
        const float nrr = x + hr * ar - hi * ai;
        const float nii = hr * ai + hi * ar;
        hr = nrr;
        hi = nii;
    }
    const size_t o = (((size_t)b * NC + c) * Hd + h) * 2;
    carry[o + 0] = hr;
    carry[o + 1] = hi;
}

// ---------------------------------------------------------------------------
// Kernel 3: prefix over chunks (in place: carry buffer becomes hstart buffer).
//   E_{-1} = 0;  hstart[c] = E_{c-1};  E_c = Apow * E_{c-1} + carry_c
// ---------------------------------------------------------------------------
__global__ void lssm_prefix_kernel(const float* __restrict__ apow,
                                   float* __restrict__ carry) {
    const int b = blockIdx.x;
    const int h = threadIdx.x;
    const float pr = apow[(b * Hd + h) * 2 + 0];
    const float pi = apow[(b * Hd + h) * 2 + 1];
    float er = 0.0f, ei = 0.0f;
#pragma unroll
    for (int c = 0; c < NC; ++c) {
        const size_t o = (((size_t)b * NC + c) * Hd + h) * 2;
        const float vr = carry[o + 0];
        const float vi = carry[o + 1];
        carry[o + 0] = er;             // now holds chunk start state
        carry[o + 1] = ei;
        const float nrr = pr * er - pi * ei + vr;
        const float nii = pr * ei + pi * er + vi;
        er = nrr;
        ei = nii;
    }
}

// ---------------------------------------------------------------------------
// Kernel 4: fused scan + WMMA GEMM.  One (batch, chunk) per block, 8 waves.
// Per 32-step tile: threads 0..127 run the diagonal scan and deposit f16
// h-state into LDS; then all 8 waves compute Y_tile(32x128) =
// HRe @ CbRe^T + HIm @ (-CbIm^T) with v_wmma_f32_16x16x32_f16.
// ---------------------------------------------------------------------------
__global__ __launch_bounds__(256) void lssm_main_kernel(
    const float* __restrict__ X,
    const float* __restrict__ abarRe,
    const float* __restrict__ abarIm,
    const float* __restrict__ hstart,
    const unsigned int* __restrict__ bfragU,
    float* __restrict__ Y) {
    __shared__ _Float16 sRe[LT][LDP];
    __shared__ _Float16 sIm[LT][LDP];

    const int c = blockIdx.x;
    const int b = blockIdx.y;
    const int tid = threadIdx.x;
    const int lane = tid & 31;
    const int wave = tid >> 5;

    float ar = 0.0f, ai = 0.0f, hr = 0.0f, hi = 0.0f;
    if (tid < Hd) {
        ar = abarRe[b * Hd + tid];
        ai = abarIm[b * Hd + tid];
        const size_t o = (((size_t)b * NC + c) * Hd + tid) * 2;
        hr = hstart[o + 0];
        hi = hstart[o + 1];
    }

    for (int tile = 0; tile < NTILES; ++tile) {
        const int t0 = c * LC + tile * LT;

        // ---- scan phase: 32 dependent steps, x prefetched into registers ----
        if (tid < Hd) {
            const float* xp = X + ((size_t)b * Ld + t0) * Hd + tid;
            float xv[LT];
#pragma unroll
            for (int lt = 0; lt < LT; ++lt) xv[lt] = xp[(size_t)lt * Hd];
#pragma unroll
            for (int lt = 0; lt < LT; ++lt) {
                const float nrr = xv[lt] + hr * ar - hi * ai;
                const float nii = hr * ai + hi * ar;
                hr = nrr;
                hi = nii;
                sRe[lt][tid] = (_Float16)hr;
                sIm[lt][tid] = (_Float16)hi;
            }
            if (tile + 1 < NTILES)
                __builtin_prefetch(xp + (size_t)LT * Hd, 0, 1);  // next tile's x
        }
        __syncthreads();

        // ---- GEMM phase: 16 output fragments, 2 per wave ----
#pragma unroll
        for (int fsel = 0; fsel < 2; ++fsel) {
            const int f = wave + fsel * 8;
            const int mt = f >> 3;      // 0..1 : which 16 rows of the time tile
            const int nb = f & 7;       // 0..7 : which 16 output columns
            v8f acc = {0.f, 0.f, 0.f, 0.f, 0.f, 0.f, 0.f, 0.f};

            const int M = mt * 16 + (lane & 15);
            const int khalf = lane >> 4;

#pragma unroll
            for (int kb = 0; kb < 4; ++kb) {
                // A fragments from LDS (16x32 f16, ISA A-operand layout)
                union { unsigned int u[8]; v16h v; } aR, aI;
#pragma unroll
                for (int d = 0; d < 8; ++d) {
                    const int k0 = kb * 32 + ((d >> 2) << 4) + (khalf << 3) + ((d & 3) << 1);
                    aR.u[d] = *(const unsigned int*)&sRe[M][k0];
                    aI.u[d] = *(const unsigned int*)&sIm[M][k0];
                }
                // B fragments from L2-resident packed buffer (two uint4 per lane)
                const unsigned int* p1 = bfragU + (size_t)((kb * 8 + nb) * 32 + lane) * 8;
                const unsigned int* p2 = bfragU + (size_t)((32 + kb * 8 + nb) * 32 + lane) * 8;
                union { uint4 q[2]; v16h v; } bR, bI;
                bR.q[0] = *(const uint4*)(p1);
                bR.q[1] = *(const uint4*)(p1 + 4);
                bI.q[0] = *(const uint4*)(p2);
                bI.q[1] = *(const uint4*)(p2 + 4);

                acc = __builtin_amdgcn_wmma_f32_16x16x32_f16(
                    false, aR.v, false, bR.v, (short)0, acc, false, false);
                acc = __builtin_amdgcn_wmma_f32_16x16x32_f16(
                    false, aI.v, false, bI.v, (short)0, acc, false, false);
            }

            // D layout: VGPR r -> M = r + 8*(lane>=16), N = lane%16
            const int rowBase = mt * 16 + khalf * 8;
            const int col = nb * 16 + (lane & 15);
            float* yp = Y + ((size_t)b * Ld + (t0 + rowBase)) * Hd + col;
#pragma unroll
            for (int r = 0; r < 8; ++r) yp[(size_t)r * Hd] = acc[r];
        }
        __syncthreads();
    }
}

// ---------------------------------------------------------------------------
// Host-side launcher. Workspace layout (bytes):
//   [0,      65536)  abarRe (H*H f32)
//   [65536, 131072)  abarIm
//   [131072,262144)  apow  (H*H complex f32)
//   [262144,327680)  bfrag (64 f16 fragments, 1 KB each)
//   [327680,2424832) carry -> hstart (B*NC*H complex f32)
// ---------------------------------------------------------------------------
extern "C" void kernel_launch(void* const* d_in, const int* in_sizes, int n_in,
                              void* d_out, int out_size, void* d_ws, size_t ws_size,
                              hipStream_t stream) {
    const float* X    = (const float*)d_in[0];
    const float* A_rt = (const float*)d_in[1];
    const float* C_rt = (const float*)d_in[2];
    float* out = (float*)d_out;

    char* ws = (char*)d_ws;
    float*    abarRe = (float*)(ws + 0);
    float*    abarIm = (float*)(ws + 65536);
    float*    apow   = (float*)(ws + 131072);
    _Float16* bfrag  = (_Float16*)(ws + 262144);
    float*    carry  = (float*)(ws + 327680);

    lssm_setup_kernel<<<dim3(Hd), dim3(Hd), 0, stream>>>(
        A_rt, C_rt, abarRe, abarIm, apow, bfrag);
    lssm_carry_kernel<<<dim3(NC, Bd), dim3(Hd), 0, stream>>>(
        X, abarRe, abarIm, carry);
    lssm_prefix_kernel<<<dim3(Bd), dim3(Hd), 0, stream>>>(apow, carry);
    lssm_main_kernel<<<dim3(NC, Bd), dim3(256), 0, stream>>>(
        X, abarRe, abarIm, carry, (const unsigned int*)bfrag, out);
}
